// DeepseekV4SparseMoeBlock_13761075216604
// MI455X (gfx1250) — compile-verified, matrix operand order checked by reading, not run on hardware
//
#include <hip/hip_runtime.h>
#include <hip/hip_bf16.h>

// ---------------- problem constants ----------------
#define E_   32
#define K_   8
#define H_   1024
#define I_   512
#define SI_  2048
#define CAP_ 2048
#define T_   4096          // B*S = 2*2048
#define LIMIT_ 7.0f
#define SCALE_ 2.5f

// ---------------- WMMA types ----------------
typedef __attribute__((ext_vector_type(16))) __bf16 v16bf;
typedef __attribute__((ext_vector_type(8)))  float  v8f;

__device__ __forceinline__ v8f wmma_bf16(v16bf a, v16bf b, v8f c) {
    // D = A(16x32 bf16) * B(32x16 bf16) + C(16x16 f32)
    return __builtin_amdgcn_wmma_f32_16x16x32_bf16(false, a, false, b, (short)0, c, false, false);
}

__device__ __forceinline__ unsigned short f2bf(float f) {
    union { float f; unsigned u; } c; c.f = f;
    unsigned u = c.u + 0x7FFFu + ((c.u >> 16) & 1u);   // round-to-nearest-even
    return (unsigned short)(u >> 16);
}

// Fragment from bf16 (ushort) memory: two 16B chunks at [koff] and [16+koff]
__device__ __forceinline__ v16bf frag_bf16(const unsigned short* p, int koff) {
    union { uint4 q[2]; v16bf v; } r;
    r.q[0] = *(const uint4*)(p + koff);
    r.q[1] = *(const uint4*)(p + 16 + koff);
    return r.v;
}

__device__ __forceinline__ float silu_f(float x) { return x / (1.0f + __expf(-x)); }

// ---------------- fp32 -> bf16 bulk convert (float4 / ushort4 vectorized) ----------------
__global__ void convert_bf16_kernel(const float* __restrict__ src,
                                    unsigned short* __restrict__ dst,
                                    long long n4) {
    long long id = (long long)blockIdx.x * blockDim.x + threadIdx.x;
    if (id < n4) {
        float4 v = ((const float4*)src)[id];
        ushort4 o;
        o.x = f2bf(v.x); o.y = f2bf(v.y); o.z = f2bf(v.z); o.w = f2bf(v.w);
        ((ushort4*)dst)[id] = o;
    }
}

__global__ void zero_counts_kernel(int* __restrict__ counts) {
    if (threadIdx.x < E_) counts[threadIdx.x] = 0;
}

// ---------------- router (one wave32 per token) ----------------
__global__ void router_kernel(const float* __restrict__ x,
                              const float* __restrict__ rw,
                              const float* __restrict__ bias,
                              int* __restrict__ counts,
                              int* __restrict__ slot_tok,
                              float* __restrict__ slot_w) {
    const int t    = blockIdx.x;
    const int lane = threadIdx.x;   // 0..31 == expert id

    const float4* xv = (const float4*)(x + (size_t)t * H_);
    const float4* wv = (const float4*)(rw + (size_t)lane * H_);
    float acc = 0.0f;
#pragma unroll 4
    for (int i = 0; i < H_ / 4; ++i) {
        float4 a = xv[i], b = wv[i];
        acc += a.x * b.x + a.y * b.y + a.z * b.z + a.w * b.w;
    }
    const float score = 1.0f / (1.0f + __expf(-acc));
    float selv = score + bias[lane];

    int   ek[K_];
    float wk[K_];
#pragma unroll
    for (int k = 0; k < K_; ++k) {
        float v = selv; int i = lane;
#pragma unroll
        for (int off = 16; off >= 1; off >>= 1) {
            float ov = __shfl_xor(v, off, 32);
            int   oi = __shfl_xor(i, off, 32);
            if (ov > v || (ov == v && oi < i)) { v = ov; i = oi; }
        }
        ek[k] = i;
        wk[k] = __shfl(score, i, 32);
        if (lane == i) selv = -3.4e38f;
    }
    float wsum = 1e-20f;
#pragma unroll
    for (int k = 0; k < K_; ++k) wsum += wk[k];

    if (lane < K_) {
        int   ee = ek[lane];
        float ww = wk[lane] / wsum * SCALE_;
        int p = atomicAdd(&counts[ee], 1);
        if (p < CAP_) {
            slot_tok[ee * CAP_ + p] = t;
            slot_w  [ee * CAP_ + p] = ww;
        }
    }
}

// ---------------- shared GEMM1: hs = silu(x@Wg^T)*(x@Wu^T), bf16 out ----------------
// grid (SI/64, T/128), block 256 (8 waves). Wave -> 32x32 patch, dual matrix (2x2x2 tiles).
__global__ void shared_gemm1_kernel(const unsigned short* __restrict__ xbf,
                                    const unsigned short* __restrict__ sg,
                                    const unsigned short* __restrict__ su,
                                    unsigned short* __restrict__ hs) {
    const int lane = threadIdx.x & 31, w = threadIdx.x >> 5;
    const int base_m = blockIdx.y * 128 + (w >> 1) * 32;
    const int base_n = blockIdx.x * 64  + (w & 1) * 32;
    const int koff = (lane >> 4) * 8;

    const unsigned short* a0p = xbf + (size_t)(base_m      + (lane & 15)) * H_;
    const unsigned short* a1p = xbf + (size_t)(base_m + 16 + (lane & 15)) * H_;
    const unsigned short* g0p = sg  + (size_t)(base_n      + (lane & 15)) * H_;
    const unsigned short* g1p = sg  + (size_t)(base_n + 16 + (lane & 15)) * H_;
    const unsigned short* u0p = su  + (size_t)(base_n      + (lane & 15)) * H_;
    const unsigned short* u1p = su  + (size_t)(base_n + 16 + (lane & 15)) * H_;

    v8f ag[2][2] = {{{}, {}}, {{}, {}}};
    v8f au[2][2] = {{{}, {}}, {{}, {}}};
#pragma unroll 2
    for (int kk = 0; kk < H_ / 32; ++kk) {
        v16bf a0 = frag_bf16(a0p + kk * 32, koff);
        v16bf a1 = frag_bf16(a1p + kk * 32, koff);
        v16bf g0 = frag_bf16(g0p + kk * 32, koff);
        v16bf g1 = frag_bf16(g1p + kk * 32, koff);
        v16bf u0 = frag_bf16(u0p + kk * 32, koff);
        v16bf u1 = frag_bf16(u1p + kk * 32, koff);
        ag[0][0] = wmma_bf16(a0, g0, ag[0][0]);
        ag[0][1] = wmma_bf16(a0, g1, ag[0][1]);
        ag[1][0] = wmma_bf16(a1, g0, ag[1][0]);
        ag[1][1] = wmma_bf16(a1, g1, ag[1][1]);
        au[0][0] = wmma_bf16(a0, u0, au[0][0]);
        au[0][1] = wmma_bf16(a0, u1, au[0][1]);
        au[1][0] = wmma_bf16(a1, u0, au[1][0]);
        au[1][1] = wmma_bf16(a1, u1, au[1][1]);
    }
#pragma unroll
    for (int i = 0; i < 2; ++i)
#pragma unroll
        for (int j = 0; j < 2; ++j)
#pragma unroll
            for (int r = 0; r < 8; ++r) {
                int m = base_m + i * 16 + (lane >> 4) * 8 + r;
                int n = base_n + j * 16 + (lane & 15);
                hs[(size_t)m * SI_ + n] = f2bf(silu_f(ag[i][j][r]) * au[i][j][r]);
            }
}

// ---------------- shared GEMM2: out = hs @ Wd^T (fp32 out) ----------------
// grid (H/256, T/128), block 256 (8 waves as 2x4). Wave -> 64x64 patch (4x4 tiles).
__global__ void shared_gemm2_kernel(const unsigned short* __restrict__ hs,
                                    const unsigned short* __restrict__ sd,
                                    float* __restrict__ out) {
    const int lane = threadIdx.x & 31, w = threadIdx.x >> 5;
    const int base_m = blockIdx.y * 128 + (w & 1) * 64;
    const int base_n = blockIdx.x * 256 + (w >> 1) * 64;
    const int koff = (lane >> 4) * 8;

    const unsigned short* ap[4];
    const unsigned short* bp[4];
#pragma unroll
    for (int i = 0; i < 4; ++i) {
        ap[i] = hs + (size_t)(base_m + i * 16 + (lane & 15)) * SI_;
        bp[i] = sd + (size_t)(base_n + i * 16 + (lane & 15)) * SI_;
    }

    v8f acc[4][4] = {{{},{},{},{}},{{},{},{},{}},{{},{},{},{}},{{},{},{},{}}};
#pragma unroll 2
    for (int kk = 0; kk < SI_ / 32; ++kk) {
        v16bf a[4], b[4];
#pragma unroll
        for (int i = 0; i < 4; ++i) a[i] = frag_bf16(ap[i] + kk * 32, koff);
#pragma unroll
        for (int j = 0; j < 4; ++j) b[j] = frag_bf16(bp[j] + kk * 32, koff);
#pragma unroll
        for (int i = 0; i < 4; ++i)
#pragma unroll
            for (int j = 0; j < 4; ++j)
                acc[i][j] = wmma_bf16(a[i], b[j], acc[i][j]);
    }
#pragma unroll
    for (int i = 0; i < 4; ++i)
#pragma unroll
        for (int j = 0; j < 4; ++j)
#pragma unroll
            for (int r = 0; r < 8; ++r) {
                int m = base_m + i * 16 + (lane >> 4) * 8 + r;
                int n = base_n + j * 16 + (lane & 15);
                out[(size_t)m * H_ + n] = acc[i][j][r];
            }
}

// ---------------- routed experts: fused gate_up -> act -> down ----------------
// grid (CAP/32, E), block 256 (8 waves). Block tile = 32 gathered token rows of one expert.
__global__ void expert_ffn_kernel(const unsigned short* __restrict__ xbf,
                                  const unsigned short* __restrict__ gup,
                                  const unsigned short* __restrict__ dwn,
                                  const int* __restrict__ counts,
                                  const int* __restrict__ slot_tok,
                                  const float* __restrict__ slot_w,
                                  float* __restrict__ out) {
    const int e    = blockIdx.y;
    const int base = blockIdx.x * 32;
    const int cnt  = counts[e] < CAP_ ? counts[e] : CAP_;
    if (base >= cnt) return;

    __shared__ unsigned short h_lds[32 * I_];   // 32 KB bf16 intermediate
    __shared__ int   s_tok[32];
    __shared__ float s_w[32];

    const int lane = threadIdx.x & 31, w = threadIdx.x >> 5;
    if (threadIdx.x < 32) {
        int slot = base + threadIdx.x;
        bool valid = slot < cnt;
        s_tok[threadIdx.x] = valid ? slot_tok[e * CAP_ + slot] : 0;
        s_w  [threadIdx.x] = valid ? slot_w  [e * CAP_ + slot] : 0.0f;
    }
    __syncthreads();

    const int koff = (lane >> 4) * 8;
    const unsigned short* a0p = xbf + (size_t)s_tok[lane & 15] * H_;
    const unsigned short* a1p = xbf + (size_t)s_tok[16 + (lane & 15)] * H_;

    // ---- phase 1: h = silu(min(g,L)) * clip(u); wave -> M=32 x N=64 dual (2x4x2 tiles) ----
    {
        const int nb = w * 64;                   // 8 waves * 64 cols = all of I=512
        const unsigned short* gp[4];
        const unsigned short* up[4];
#pragma unroll
        for (int j = 0; j < 4; ++j) {
            gp[j] = gup + ((size_t)e * 2 * I_ + nb + j * 16 + (lane & 15)) * H_;
            up[j] = gup + ((size_t)e * 2 * I_ + I_ + nb + j * 16 + (lane & 15)) * H_;
        }
        v8f ag[2][4] = {{{},{},{},{}},{{},{},{},{}}};
        v8f au[2][4] = {{{},{},{},{}},{{},{},{},{}}};
#pragma unroll 2
        for (int kk = 0; kk < H_ / 32; ++kk) {
            v16bf a0 = frag_bf16(a0p + kk * 32, koff);
            v16bf a1 = frag_bf16(a1p + kk * 32, koff);
            v16bf g[4], u[4];
#pragma unroll
            for (int j = 0; j < 4; ++j) g[j] = frag_bf16(gp[j] + kk * 32, koff);
#pragma unroll
            for (int j = 0; j < 4; ++j) u[j] = frag_bf16(up[j] + kk * 32, koff);
#pragma unroll
            for (int j = 0; j < 4; ++j) {
                ag[0][j] = wmma_bf16(a0, g[j], ag[0][j]);
                ag[1][j] = wmma_bf16(a1, g[j], ag[1][j]);
                au[0][j] = wmma_bf16(a0, u[j], au[0][j]);
                au[1][j] = wmma_bf16(a1, u[j], au[1][j]);
            }
        }
#pragma unroll
        for (int i = 0; i < 2; ++i)
#pragma unroll
            for (int j = 0; j < 4; ++j)
#pragma unroll
                for (int r = 0; r < 8; ++r) {
                    int m = i * 16 + (lane >> 4) * 8 + r;
                    int n = nb + j * 16 + (lane & 15);
                    float g = fminf(ag[i][j][r], LIMIT_);
                    float u = fminf(fmaxf(au[i][j][r], -LIMIT_), LIMIT_);
                    h_lds[m * I_ + n] = f2bf(silu_f(g) * u);
                }
    }
    __syncthreads();

    // ---- phase 2: ye = h @ down[e]^T; wave -> M=32 x N=64 (2x4 tiles); atomic-add out ----
    const unsigned short* ha0 = h_lds + (lane & 15) * I_;
    const unsigned short* ha1 = h_lds + (16 + (lane & 15)) * I_;
    for (int cpos = w; cpos < H_ / 64; cpos += 8) {   // 16 col blocks / 8 waves = 2 iters
        const int nb = cpos * 64;
        const unsigned short* bp[4];
#pragma unroll
        for (int j = 0; j < 4; ++j)
            bp[j] = dwn + ((size_t)e * H_ + nb + j * 16 + (lane & 15)) * I_;

        v8f acc[2][4] = {{{},{},{},{}},{{},{},{},{}}};
#pragma unroll 2
        for (int kk = 0; kk < I_ / 32; ++kk) {
            v16bf a0 = frag_bf16(ha0 + kk * 32, koff);
            v16bf a1 = frag_bf16(ha1 + kk * 32, koff);
            v16bf b[4];
#pragma unroll
            for (int j = 0; j < 4; ++j) b[j] = frag_bf16(bp[j] + kk * 32, koff);
#pragma unroll
            for (int j = 0; j < 4; ++j) {
                acc[0][j] = wmma_bf16(a0, b[j], acc[0][j]);
                acc[1][j] = wmma_bf16(a1, b[j], acc[1][j]);
            }
        }
#pragma unroll
        for (int i = 0; i < 2; ++i)
#pragma unroll
            for (int j = 0; j < 4; ++j)
#pragma unroll
                for (int r = 0; r < 8; ++r) {
                    int m = i * 16 + (lane >> 4) * 8 + r;
                    if (base + m < cnt) {
                        int n = nb + j * 16 + (lane & 15);
                        atomicAdd(&out[(size_t)s_tok[m] * H_ + n], acc[i][j][r] * s_w[m]);
                    }
                }
    }
}

// ---------------- host side ----------------
extern "C" void kernel_launch(void* const* d_in, const int* in_sizes, int n_in,
                              void* d_out, int out_size, void* d_ws, size_t ws_size,
                              hipStream_t stream) {
    (void)in_sizes; (void)n_in; (void)out_size; (void)ws_size;
    const float* x    = (const float*)d_in[0];
    const float* rw   = (const float*)d_in[1];
    const float* bias = (const float*)d_in[2];
    const float* gup  = (const float*)d_in[3];
    const float* dwn  = (const float*)d_in[4];
    const float* sg   = (const float*)d_in[5];
    const float* su   = (const float*)d_in[6];
    const float* sd   = (const float*)d_in[7];
    float* out = (float*)d_out;

    // workspace layout (~138 MB)
    char* w = (char*)d_ws;
    size_t o = 0;
    int*   counts   = (int*)(w + o);              o += 4096;
    int*   slot_tok = (int*)(w + o);              o += (size_t)E_ * CAP_ * 4;
    float* slot_w   = (float*)(w + o);            o += (size_t)E_ * CAP_ * 4;
    unsigned short* xbf   = (unsigned short*)(w + o); o += (size_t)T_ * H_ * 2;
    unsigned short* hsbuf = (unsigned short*)(w + o); o += (size_t)T_ * SI_ * 2;
    unsigned short* gupbf = (unsigned short*)(w + o); o += (size_t)E_ * 2 * I_ * H_ * 2;
    unsigned short* dwnbf = (unsigned short*)(w + o); o += (size_t)E_ * H_ * I_ * 2;
    unsigned short* sgbf  = (unsigned short*)(w + o); o += (size_t)SI_ * H_ * 2;
    unsigned short* subf  = (unsigned short*)(w + o); o += (size_t)SI_ * H_ * 2;
    unsigned short* sdbf  = (unsigned short*)(w + o); o += (size_t)H_ * SI_ * 2;

    auto cvt = [&](const float* s, unsigned short* d, long long n) {
        long long n4 = n / 4;
        convert_bf16_kernel<<<(unsigned)((n4 + 255) / 256), 256, 0, stream>>>(s, d, n4);
    };
    cvt(x,   xbf,   (long long)T_ * H_);
    cvt(gup, gupbf, (long long)E_ * 2 * I_ * H_);
    cvt(dwn, dwnbf, (long long)E_ * H_ * I_);
    cvt(sg,  sgbf,  (long long)SI_ * H_);
    cvt(su,  subf,  (long long)SI_ * H_);
    cvt(sd,  sdbf,  (long long)H_ * SI_);
    zero_counts_kernel<<<1, 32, 0, stream>>>(counts);

    router_kernel<<<T_, 32, 0, stream>>>(x, rw, bias, counts, slot_tok, slot_w);
    shared_gemm1_kernel<<<dim3(SI_ / 64, T_ / 128), 256, 0, stream>>>(xbf, sgbf, subf, hsbuf);
    shared_gemm2_kernel<<<dim3(H_ / 256, T_ / 128), 256, 0, stream>>>(hsbuf, sdbf, out);
    expert_ffn_kernel<<<dim3(CAP_ / 32, E_), 256, 0, stream>>>(xbf, gupbf, dwnbf, counts,
                                                               slot_tok, slot_w, out);
}
